// DFConv3x3_14628658610323
// MI455X (gfx1250) — compile-verified, matrix operand order checked by reading, not run on hardware
//
#include <hip/hip_runtime.h>
#include <hip/hip_bf16.h>
#include <stdint.h>

// Problem constants
#define NB   4
#define CIN  256
#define COUT 256
#define HH   96
#define WW   96
#define HWSZ (HH*WW)          // 9216
#define K9   9
#define KDIM (CIN*K9)         // 2304
#define EPSV 1e-5f

typedef __attribute__((ext_vector_type(16))) __bf16 v16bf;
typedef __attribute__((ext_vector_type(8)))  float  v8f;

static __device__ __forceinline__ __bf16 f2bf(float f) {
    // round-to-nearest-even f32 -> bf16
    union { float f; uint32_t u; } in; in.f = f;
    uint32_t u = in.u + 0x7FFFu + ((in.u >> 16) & 1u);
    uint16_t s = (uint16_t)(u >> 16);
    union { uint16_t s; __bf16 b; } out; out.s = s;
    return out.b;
}

// ---------------------------------------------------------------------------
// Kernel 1: repack conv weights  w[o][c][ky][kx] (f32) -> wT[o][t*CIN + c] (bf16)
// K permuted tap-major so each 32-wide K block = 32 channels of a single tap.
// ---------------------------------------------------------------------------
__global__ __launch_bounds__(256) void repack_w_kernel(const float* __restrict__ w,
                                                       __bf16* __restrict__ wT) {
    int idx = blockIdx.x * 256 + threadIdx.x;       // 0 .. COUT*KDIM-1
    if (idx >= COUT * KDIM) return;
    int o = idx / KDIM;
    int r = idx % KDIM;
    int t = r / CIN;
    int c = r % CIN;
    wT[idx] = f2bf(w[(size_t)o * KDIM + c * K9 + t]);
}

// ---------------------------------------------------------------------------
// Kernel 2: BN coefficients  scale = gamma*rsqrt(var+eps), shift = beta - mean*scale
// ---------------------------------------------------------------------------
__global__ void bn_coeff_kernel(const float* __restrict__ gamma, const float* __restrict__ beta,
                                const float* __restrict__ mean,  const float* __restrict__ var,
                                float* __restrict__ bnscale, float* __restrict__ bnshift) {
    int o = threadIdx.x;
    if (o < COUT) {
        float inv = gamma[o] * rsqrtf(var[o] + EPSV);
        bnscale[o] = inv;
        bnshift[o] = beta[o] - mean[o] * inv;
    }
}

// ---------------------------------------------------------------------------
// Kernel 3: offset conv (27 output channels), sigmoid fused on mask channels.
// One thread per output pixel; weights staged in LDS in 32-channel chunks
// ([cc][j][d] with d fast -> same-address LDS broadcast reads).
// om layout: [n][27][H*W], channels 0..8=off_x, 9..17=off_y, 18..26=sigmoid(mask)
// ---------------------------------------------------------------------------
__global__ __launch_bounds__(256) void offset_conv_kernel(const float* __restrict__ x,
                                                          const float* __restrict__ w_off,
                                                          const float* __restrict__ b_off,
                                                          float* __restrict__ om) {
    __shared__ float wl[32 * 9 * 27];                // 31104 B
    const int tid = threadIdx.x;
    const int pix = blockIdx.x * 256 + tid;          // 0..36863
    const int n  = pix / HWSZ;
    const int hw = pix % HWSZ;
    const int h  = hw / WW, w = hw % WW;

    float acc[27];
#pragma unroll
    for (int d = 0; d < 27; ++d) acc[d] = b_off[d];

    for (int c0 = 0; c0 < CIN; c0 += 32) {
        __syncthreads();
        for (int e = tid; e < 32 * 9 * 27; e += 256) {
            int d  = e % 27;
            int r  = e / 27;
            int j  = r % 9;
            int cc = r / 9;
            wl[e] = w_off[(size_t)(d * CIN + c0 + cc) * 9 + j];
        }
        __syncthreads();
        for (int cc = 0; cc < 32; ++cc) {
            const float* xp = x + (size_t)(n * CIN + c0 + cc) * HWSZ;
            float xv[9];
#pragma unroll
            for (int j = 0; j < 9; ++j) {
                int yy = h + j / 3 - 1;
                int xx = w + j % 3 - 1;
                bool ok = (yy >= 0) && (yy < HH) && (xx >= 0) && (xx < WW);
                xv[j] = ok ? xp[yy * WW + xx] : 0.f;
            }
            const float* wp = &wl[cc * 9 * 27];
#pragma unroll
            for (int j = 0; j < 9; ++j)
#pragma unroll
                for (int d = 0; d < 27; ++d)
                    acc[d] = fmaf(xv[j], wp[j * 27 + d], acc[d]);
        }
    }
#pragma unroll
    for (int d = 0; d < 27; ++d) {
        float v = acc[d];
        if (d >= 18) v = 1.0f / (1.0f + __expf(-v));     // sigmoid for mask
        om[((size_t)n * 27 + d) * HWSZ + hw] = v;
    }
}

// ---------------------------------------------------------------------------
// Kernel 4: fused deformable-im2col + bf16 WMMA GEMM + BN + ReLU
// Grid: (N*H*W/64), block 256 (8 waves). Tile: 256 M (full COUT) x 64 pixels.
// Each wave: 2 M-groups x 4 N-subtiles = 8 accumulators, 8 WMMAs per K-block.
// B tile is double-buffered: one barrier per K-block; next block's gathers
// overlap with current block's WMMAs.
// ---------------------------------------------------------------------------
__global__ __launch_bounds__(256) void dcn_gemm_kernel(const float* __restrict__ x,
                                                       const float* __restrict__ om,
                                                       const __bf16* __restrict__ wT,
                                                       const float* __restrict__ bnscale,
                                                       const float* __restrict__ bnshift,
                                                       float* __restrict__ out) {
    __shared__ int    soff[64 * 9 * 4];      // 9216 B : 4 gather offsets per (pixel,tap)
    __shared__ float  swgt[64 * 9 * 4];      // 9216 B : 4 folded weights (bilinear*mask*valid)
    __shared__ __bf16 Bt[2][64 * 32];        // 8192 B : double-buffered B tile [pixel][k]

    const int tid  = threadIdx.x;
    const int tile = blockIdx.x;             // 0..575 (64-pixel tiles, all in one image)
    const int pix0 = tile * 64;
    const int n    = pix0 / HWSZ;
    const int hw0  = pix0 % HWSZ;

    // ---- phase 1: per-(pixel,tap) sample records -------------------------
    for (int e = tid; e < 64 * 9; e += 256) {
        int p = e / 9, t = e % 9;
        int hw = hw0 + p;
        int h = hw / WW, w = hw % WW;
        const float* omp = om + (size_t)n * 27 * HWSZ + hw;
        float ox = omp[(size_t)t * HWSZ];
        float oy = omp[(size_t)(9 + t) * HWSZ];
        float mk = omp[(size_t)(18 + t) * HWSZ];
        float sy = oy + (float)(t / 3 - 1) + (float)h;
        float sx = ox + (float)(t % 3 - 1) + (float)w;
        float y0f = floorf(sy), x0f = floorf(sx);
        float wy1 = sy - y0f, wx1 = sx - x0f;
        float wy0 = 1.f - wy1, wx0 = 1.f - wx1;
        int y0 = (int)y0f, x0 = (int)x0f;
#pragma unroll
        for (int j = 0; j < 4; ++j) {
            int yi = y0 + (j >> 1);
            int xi = x0 + (j & 1);
            float wj = ((j >> 1) ? wy1 : wy0) * ((j & 1) ? wx1 : wx0) * mk;
            bool ok = (yi >= 0) && (yi < HH) && (xi >= 0) && (xi < WW);
            soff[e * 4 + j] = ok ? (yi * WW + xi) : 0;
            swgt[e * 4 + j] = ok ? wj : 0.f;
        }
    }

    const int lane = tid & 31;
    const int wv   = tid >> 5;               // wave 0..7
    const int half = lane >> 4;
    const int l15  = lane & 15;

    const int bp  = tid >> 2;                // pixel this thread builds (0..63)
    const int bkg = (tid & 3) * 8;           // k-group base (0,8,16,24)

    // build: gather 8 channels x 4 corners, fold into 8 bf16, one b128 LDS store
    auto buildB = [&](int kb, __bf16* dst) {
        const int t  = kb >> 3;
        const int c0 = (kb & 7) * 32;
        const int e  = bp * 9 + t;
        const int o0 = soff[e * 4 + 0], o1 = soff[e * 4 + 1];
        const int o2 = soff[e * 4 + 2], o3 = soff[e * 4 + 3];
        const float g0 = swgt[e * 4 + 0], g1 = swgt[e * 4 + 1];
        const float g2 = swgt[e * 4 + 2], g3 = swgt[e * 4 + 3];
        const float* xp = x + (size_t)(n * CIN + c0 + bkg) * HWSZ;
        union { __bf16 h[8]; uint4 q; } vals;
#pragma unroll
        for (int u = 0; u < 8; ++u) {
            const float* xc = xp + (size_t)u * HWSZ;
            float v = g0 * xc[o0] + g1 * xc[o1] + g2 * xc[o2] + g3 * xc[o3];
            vals.h[u] = f2bf(v);
        }
        *reinterpret_cast<uint4*>(&dst[bp * 32 + bkg]) = vals.q;
    };

    v8f acc[2][4] = {};                      // [M-group][N-subtile] 16x16 f32 tiles

    __syncthreads();                         // soff/swgt ready
    buildB(0, Bt[0]);                        // pre-build first K-block

    union AB { v16bf v; uint4 q[2]; };

    for (int i = 0; i < 72; ++i) {           // 9 taps x 8 channel-chunks
        __syncthreads();                     // Bt[i&1] ready; Bt[(i+1)&1] free
        const int buf = i & 1;
        if (i + 1 < 72) buildB(i + 1, Bt[buf ^ 1]);

        const int t  = i >> 3;
        const int c0 = (i & 7) * 32;

        // A fragments (documented 16-bit 16x32 layout):
        // element e <-> K = (e&7) + half*8 + (e>>3)*16  => two contiguous b128 loads
        AB A[2];
#pragma unroll
        for (int g = 0; g < 2; ++g) {
            const int m = g * 128 + wv * 16 + l15;
            const __bf16* ap = wT + (size_t)m * KDIM + t * CIN + c0 + half * 8;
            A[g].q[0] = *reinterpret_cast<const uint4*>(ap);
            A[g].q[1] = *reinterpret_cast<const uint4*>(ap + 16);
        }

        // 4 B fragments, each reused for 2 WMMAs
#pragma unroll
        for (int s = 0; s < 4; ++s) {
            AB B;
            const __bf16* bq = &Bt[buf][(s * 16 + l15) * 32 + half * 8];
            B.q[0] = *reinterpret_cast<const uint4*>(bq);
            B.q[1] = *reinterpret_cast<const uint4*>(bq + 16);
#pragma unroll
            for (int g = 0; g < 2; ++g) {
                acc[g][s] = __builtin_amdgcn_wmma_f32_16x16x32_bf16(
                    false, A[g].v, false, B.v, (short)0, acc[g][s], false, false);
            }
        }
    }

    // ---- epilogue: BN + ReLU, C/D layout: M = g*128 + wv*16 + half*8 + v, N = l15
    float sc[2][8], sh[2][8];
#pragma unroll
    for (int g = 0; g < 2; ++g)
#pragma unroll
        for (int v = 0; v < 8; ++v) {
            int m = g * 128 + wv * 16 + half * 8 + v;
            sc[g][v] = bnscale[m];
            sh[g][v] = bnshift[m];
        }
#pragma unroll
    for (int s = 0; s < 4; ++s) {
        int hw = hw0 + s * 16 + l15;
#pragma unroll
        for (int g = 0; g < 2; ++g)
#pragma unroll
            for (int v = 0; v < 8; ++v) {
                int m = g * 128 + wv * 16 + half * 8 + v;
                float r = acc[g][s][v] * sc[g][v] + sh[g][v];
                r = fmaxf(r, 0.f);
                out[((size_t)n * COUT + m) * HWSZ + hw] = r;
            }
    }
}

// ---------------------------------------------------------------------------
extern "C" void kernel_launch(void* const* d_in, const int* in_sizes, int n_in,
                              void* d_out, int out_size, void* d_ws, size_t ws_size,
                              hipStream_t stream) {
    const float* x     = (const float*)d_in[0];
    const float* w_off = (const float*)d_in[1];
    const float* b_off = (const float*)d_in[2];
    const float* w     = (const float*)d_in[3];
    const float* gamma = (const float*)d_in[4];
    const float* beta  = (const float*)d_in[5];
    const float* mean  = (const float*)d_in[6];
    const float* var   = (const float*)d_in[7];
    float* out = (float*)d_out;

    char* ws = (char*)d_ws;
    float*  om      = (float*)ws;                                  // 4*27*9216 f32 = 3,981,312 B
    __bf16* wT      = (__bf16*)(ws + 3981312);                     // 256*2304 bf16 = 1,179,648 B
    float*  bnscale = (float*)(ws + 3981312 + 1179648);            // 256 f32
    float*  bnshift = bnscale + 256;

    // 1) weight repack (f32 -> bf16, tap-major K)
    repack_w_kernel<<<(COUT * KDIM + 255) / 256, 256, 0, stream>>>(w, wT);
    // 2) BN coefficients
    bn_coeff_kernel<<<1, 256, 0, stream>>>(gamma, beta, mean, var, bnscale, bnshift);
    // 3) offset conv (27 ch) + sigmoid on mask channels
    offset_conv_kernel<<<(NB * HWSZ) / 256, 256, 0, stream>>>(x, w_off, b_off, om);
    // 4) fused deformable im2col + WMMA GEMM + BN + ReLU (full-COUT M tile)
    dcn_gemm_kernel<<<(NB * HWSZ) / 64, 256, 0, stream>>>(x, om, wT, bnscale, bnshift, out);
}
